// update_p_55387898250016
// MI455X (gfx1250) — compile-verified, matrix operand order checked by reading, not run on hardware
//
#include <hip/hip_runtime.h>
#include <hip/hip_bf16.h>
#include <cstdint>
#include <cstddef>

// ---------------------------------------------------------------------------
// Types for CDNA5 WMMA (wave32): bf16 16x16x32 -> f32 accum
// ---------------------------------------------------------------------------
typedef __attribute__((ext_vector_type(16))) __bf16 bf16x16;
typedef __attribute__((ext_vector_type(8)))  float  floatx8;
typedef __attribute__((ext_vector_type(4)))  unsigned int u32x4;
typedef __attribute__((ext_vector_type(8)))  int          i32x8;
typedef __attribute__((ext_vector_type(4)))  int          i32x4;

union FragU { bf16x16 v; uint4 q[2]; };

__device__ __forceinline__ float bf2f(unsigned short h) {
  unsigned int u = ((unsigned int)h) << 16;
  float f; __builtin_memcpy(&f, &u, 4); return f;
}
__device__ __forceinline__ unsigned short f2bf(float f) {
  unsigned int u; __builtin_memcpy(&u, &f, 4);
  u += 0x7FFFu + ((u >> 16) & 1u);           // round-to-nearest-even
  return (unsigned short)(u >> 16);
}
__device__ __forceinline__ float silu_f(float v) {
  return v / (1.0f + __expf(-v));
}

// ---------------------------------------------------------------------------
// TDM: 1-D fp32 tile copy Global -> LDS via Tensor Data Mover.
// D# group0: count=1, lds_addr, global_addr[56:0], type=2 ("image").
// D# group1: data_size=4B (code 2), tensor_dim0 = tile_dim0 = nElems,
//            tensor_dim1 = tile_dim1 = 1, tensor_dim0_stride = nElems.
// Issued by one wave; completion via TENSORcnt (s_wait_tensorcnt 0).
// ---------------------------------------------------------------------------
#if __has_builtin(__builtin_amdgcn_tensor_load_to_lds) && \
    __has_builtin(__builtin_amdgcn_s_wait_tensorcnt)
#define HAVE_TDM 1
#endif

#ifdef HAVE_TDM
__device__ __forceinline__ void tdm_load_1d_f32(const float* gsrc,
                                                unsigned int lds_byte_addr,
                                                int nElems) {
  unsigned long long ga = (unsigned long long)(uintptr_t)gsrc;
  u32x4 g0;
  g0[0] = 1u;                                            // count=1 (valid user D#)
  g0[1] = lds_byte_addr;                                 // lds_addr[31:0]
  g0[2] = (unsigned int)(ga & 0xFFFFFFFFu);              // global_addr[31:0]
  g0[3] = (unsigned int)((ga >> 32) & 0x01FFFFFFu)       // global_addr[56:32]
          | (2u << 30);                                  // type=2
  unsigned int n = (unsigned int)nElems;
  i32x8 g1;
  g1[0] = (int)(2u << 16);                               // data_size = 4 bytes
  g1[1] = (int)((n & 0xFFFFu) << 16);                    // tensor_dim0[15:0] @ bits63:48
  g1[2] = (int)((n >> 16) & 0xFFFFu) | (1 << 16);        // tensor_dim0 hi | tensor_dim1=1
  g1[3] = (int)((n & 0xFFFFu) << 16);                    // tile_dim0 @ bits127:112
  g1[4] = 1;                                             // tile_dim1 = 1
  g1[5] = (int)n;                                        // tensor_dim0_stride lo
  g1[6] = 0;
  g1[7] = 0;
  i32x4 z4 = {0, 0, 0, 0};
#if defined(__clang_major__) && (__clang_major__ >= 23)
  i32x8 z8 = {0, 0, 0, 0, 0, 0, 0, 0};
  __builtin_amdgcn_tensor_load_to_lds(g0, g1, z4, z4, z8, 0);
#else
  __builtin_amdgcn_tensor_load_to_lds(g0, g1, z4, z4, 0);
#endif
}
#endif

// ---------------------------------------------------------------------------
// Wave-level GEMM: OUT[16 x NT*16] += SRC[16 x K] * W[NT*16 x K]^T
// SRC: bf16 tile in LDS, row-major, stride srcStride (halfs)
// W:   bf16 weight in global, row-major [outDim x K] (out row n = output col n)
// A-frag layout (16-bit A 16x32): lane m=lane&15 -> row, kh=lane>>4 -> K-half;
// VGPR0..3 hold K = kh*8+0..7, VGPR4..7 hold K = 16+kh*8+0..7 (two b128 loads).
// B mirrors with N on lane&15 (W row-major => contiguous b128 loads).
// C layout: lane&15 -> N, acc[r] -> M = r + 8*(lane>>4).
// ---------------------------------------------------------------------------
template<int K, int NT>
__device__ __forceinline__ void wmma_mm(const unsigned short* __restrict__ src,
                                        int srcStride,
                                        const unsigned short* __restrict__ W,
                                        int rowBase, int colBase0,
                                        floatx8* acc) {
  const int lane = threadIdx.x & 31;
  const int m = lane & 15, kh = lane >> 4;
  floatx8 zero = {0.f, 0.f, 0.f, 0.f, 0.f, 0.f, 0.f, 0.f};
#pragma unroll
  for (int t = 0; t < NT; ++t) acc[t] = zero;
  __builtin_prefetch(W + (size_t)colBase0 * K, 0, 1);   // global_prefetch_b8
#pragma unroll
  for (int kc = 0; kc < K; kc += 32) {
    FragU a;
    const unsigned short* pa = src + (rowBase + m) * srcStride + kc + kh * 8;
    a.q[0] = *(const uint4*)(pa);
    a.q[1] = *(const uint4*)(pa + 16);
#pragma unroll
    for (int t = 0; t < NT; ++t) {
      FragU b;
      const unsigned short* pb = W + (size_t)(colBase0 + t * 16 + m) * K + kc + kh * 8;
      b.q[0] = *(const uint4*)(pb);
      b.q[1] = *(const uint4*)(pb + 16);
      acc[t] = __builtin_amdgcn_wmma_f32_16x16x32_bf16(
          false, a.v, false, b.v, (short)0, acc[t], false, false);
    }
  }
}

// ---------------------------------------------------------------------------
// fp32 -> bf16 weight conversion
// ---------------------------------------------------------------------------
__global__ void k_f2bf(const float* __restrict__ src, unsigned short* __restrict__ dst, int n) {
  int i = blockIdx.x * 256 + threadIdx.x;
  if (i < n) dst[i] = f2bf(src[i]);
}

// ---------------------------------------------------------------------------
// Stage 1: x_p2 = silu(x1@Wp2^T+b), xp4pre = silu((silu(x1@Wp4^T+b)*sbf)@Wdown^T)
// 64 edges per block, 256 threads (8 waves).
// ---------------------------------------------------------------------------
__global__ __launch_bounds__(256) void k_stage1(
    const float* __restrict__ x1, const float* __restrict__ sbf0,
    const float* __restrict__ w_sbf1, const float* __restrict__ w_sbf2,
    const float* __restrict__ b_p2, const float* __restrict__ b_p4,
    const unsigned short* __restrict__ Wp2, const unsigned short* __restrict__ Wp4,
    const unsigned short* __restrict__ Wdown,
    unsigned short* __restrict__ xp2, unsigned short* __restrict__ xp4) {
  __shared__ __align__(16) unsigned short sX[64 * 128];
  __shared__ __align__(16) unsigned short sT[64 * 128];
  __shared__ __align__(16) float sS0[64 * 42];
  __shared__ float sS8[64 * 8];

  const int tid = threadIdx.x;
  const int eBase = blockIdx.x * 64;

#ifdef HAVE_TDM
  // TDM: async copy sbf0 tile (fp32, contiguous) into LDS; overlap with x1 cvt.
  if (tid == 0) {
    tdm_load_1d_f32(sbf0 + (size_t)eBase * 42,
                    (unsigned int)(uintptr_t)(&sS0[0]), 64 * 42);
  }
#endif
  for (int idx = tid; idx < 64 * 128; idx += 256)
    sX[idx] = f2bf(x1[(size_t)eBase * 128 + idx]);
#ifdef HAVE_TDM
  if (tid == 0) __builtin_amdgcn_s_wait_tensorcnt(0);
#else
  for (int idx = tid; idx < 64 * 42; idx += 256)
    sS0[idx] = sbf0[(size_t)eBase * 42 + idx];
#endif
  __syncthreads();

  const int wave = tid >> 5, lane = tid & 31;
  const int rowBase = (wave & 3) * 16, colBase0 = (wave >> 2) * 64;
  const int m = lane & 15, rh = (lane >> 4) << 3;
  floatx8 acc[4];

  // x_p2 -> global ws (bf16)
  wmma_mm<128, 4>(sX, 128, Wp2, rowBase, colBase0, acc);
#pragma unroll
  for (int t = 0; t < 4; ++t) {
    int col = colBase0 + t * 16 + m;
    float bc = b_p2[col];
#pragma unroll
    for (int r = 0; r < 8; ++r) {
      int row = rowBase + r + rh;
      xp2[(size_t)(eBase + row) * 128 + col] = f2bf(silu_f(acc[t][r] + bc));
    }
  }

  // x_p4 -> LDS sT
  wmma_mm<128, 4>(sX, 128, Wp4, rowBase, colBase0, acc);
#pragma unroll
  for (int t = 0; t < 4; ++t) {
    int col = colBase0 + t * 16 + m;
    float bc = b_p4[col];
#pragma unroll
    for (int r = 0; r < 8; ++r) {
      int row = rowBase + r + rh;
      sT[row * 128 + col] = f2bf(silu_f(acc[t][r] + bc));
    }
  }
  __syncthreads();

  // sbf basis: s8 = sbf0_tile @ w_sbf1^T  (42 -> 8)
  for (int q = tid; q < 64 * 8; q += 256) {
    int e = q >> 3, j = q & 7;
    const float* sr = sS0 + e * 42;
    const float* wr = w_sbf1 + j * 42;
    float s = 0.f;
    for (int k = 0; k < 42; ++k) s += sr[k] * wr[k];
    sS8[q] = s;
  }
  __syncthreads();

  // sbf = s8 @ w_sbf2^T (8 -> 128); sT *= sbf
  for (int idx = tid; idx < 64 * 128; idx += 256) {
    int e = idx >> 7, c = idx & 127;
    float s = 0.f;
#pragma unroll
    for (int j = 0; j < 8; ++j) s += sS8[e * 8 + j] * w_sbf2[c * 8 + j];
    sT[idx] = f2bf(bf2f(sT[idx]) * s);
  }
  __syncthreads();

  // down projection: [64x128] @ [64x128]^T -> [64x64]
  floatx8 acc2[2];
  const int colD0 = (wave >> 2) * 32;
  wmma_mm<128, 2>(sT, 128, Wdown, rowBase, colD0, acc2);
#pragma unroll
  for (int t = 0; t < 2; ++t) {
    int col = colD0 + t * 16 + m;
#pragma unroll
    for (int r = 0; r < 8; ++r) {
      int row = rowBase + r + rh;
      xp4[(size_t)(eBase + row) * 64 + col] = f2bf(silu_f(acc2[t][r]));
    }
  }
}

// ---------------------------------------------------------------------------
// Stage 2: gather + up-proj + residual chain + outputs.
// ---------------------------------------------------------------------------
__global__ __launch_bounds__(256) void k_stage2(
    const float* __restrict__ x1, const float* __restrict__ sbf0,
    const float* __restrict__ ta, const int* __restrict__ pidx,
    const float* __restrict__ w_t1, const float* __restrict__ w_t2,
    const unsigned short* __restrict__ xp2, const unsigned short* __restrict__ xp4,
    const unsigned short* __restrict__ Wup,
    const unsigned short* __restrict__ Wrb1, const unsigned short* __restrict__ Wrb2,
    const unsigned short* __restrict__ Wlin,
    const unsigned short* __restrict__ Wra1, const unsigned short* __restrict__ Wra2,
    const float* __restrict__ rb_b1, const float* __restrict__ rb_b2,
    const float* __restrict__ b_lin,
    const float* __restrict__ ra_b1, const float* __restrict__ ra_b2,
    const float* __restrict__ w_sbf,
    float* __restrict__ p1out, float* __restrict__ p2out) {
  __shared__ __align__(16) unsigned short sX[64 * 128];
  __shared__ __align__(16) unsigned short sT[64 * 128];
  __shared__ __align__(16) float sS0[64 * 42];
  __shared__ float sTa[64 * 64];
  __shared__ float sT8[64 * 8];

  const int tid = threadIdx.x;
  const int eBase = blockIdx.x * 64;

#ifdef HAVE_TDM
  // TDM: async copy sbf0 tile; overlap with the heavy ta basis contraction.
  if (tid == 0) {
    tdm_load_1d_f32(sbf0 + (size_t)eBase * 42,
                    (unsigned int)(uintptr_t)(&sS0[0]), 64 * 42);
  }
#else
  for (int idx = tid; idx < 64 * 42; idx += 256)
    sS0[idx] = sbf0[(size_t)eBase * 42 + idx];
#endif

  // ta basis: t8 = ta_tile @ w_t1^T  (294 -> 8)
  for (int q = tid; q < 64 * 8; q += 256) {
    int e = q >> 3, j = q & 7;
    const float* tr = ta + (size_t)(eBase + e) * 294;
    const float* wr = w_t1 + j * 294;
    float s = 0.f;
    for (int k = 0; k < 294; ++k) s += tr[k] * wr[k];
    sT8[q] = s;
  }
#ifdef HAVE_TDM
  if (tid == 0) __builtin_amdgcn_s_wait_tensorcnt(0);
#endif
  __syncthreads();

  // taI = t8 @ w_t2^T (8 -> 64)
  for (int idx = tid; idx < 64 * 64; idx += 256) {
    int e = idx >> 6, i = idx & 63;
    float s = 0.f;
#pragma unroll
    for (int j = 0; j < 8; ++j) s += sT8[e * 8 + j] * w_t2[i * 8 + j];
    sTa[idx] = s;
  }
  __syncthreads();

  // gather x_p4pre[pidx] * taI -> sT (bf16, [64 x 64], stride 64)
  for (int idx = tid; idx < 64 * 64; idx += 256) {
    int e = idx >> 6, i = idx & 63;
    int g = pidx[eBase + e];
    sT[e * 64 + i] = f2bf(bf2f(xp4[(size_t)g * 64 + i]) * sTa[idx]);
  }
  __syncthreads();

  const int wave = tid >> 5, lane = tid & 31;
  const int rowBase = (wave & 3) * 16, colBase0 = (wave >> 2) * 64;
  const int m = lane & 15, rh = (lane >> 4) << 3;
  floatx8 acc[4];

  // up projection (K=64, no bias); p1 = x_p2 + silu(...)  -> sX
  wmma_mm<64, 4>(sT, 64, Wup, rowBase, colBase0, acc);
#pragma unroll
  for (int t = 0; t < 4; ++t) {
    int col = colBase0 + t * 16 + m;
#pragma unroll
    for (int r = 0; r < 8; ++r) {
      int row = rowBase + r + rh;
      float v = silu_f(acc[t][r]) + bf2f(xp2[(size_t)(eBase + row) * 128 + col]);
      sX[row * 128 + col] = f2bf(v);
    }
  }
  __syncthreads();

  // residual block (before skip): sT = silu(sX@W1+b1); sX += silu(sT@W2+b2)
  wmma_mm<128, 4>(sX, 128, Wrb1, rowBase, colBase0, acc);
#pragma unroll
  for (int t = 0; t < 4; ++t) {
    int col = colBase0 + t * 16 + m; float bc = rb_b1[col];
#pragma unroll
    for (int r = 0; r < 8; ++r) {
      int row = rowBase + r + rh;
      sT[row * 128 + col] = f2bf(silu_f(acc[t][r] + bc));
    }
  }
  __syncthreads();
  wmma_mm<128, 4>(sT, 128, Wrb2, rowBase, colBase0, acc);
#pragma unroll
  for (int t = 0; t < 4; ++t) {
    int col = colBase0 + t * 16 + m; float bc = rb_b2[col];
#pragma unroll
    for (int r = 0; r < 8; ++r) {
      int row = rowBase + r + rh;
      int p = row * 128 + col;
      sX[p] = f2bf(bf2f(sX[p]) + silu_f(acc[t][r] + bc));
    }
  }
  __syncthreads();

  // lin: sT = silu(sX@Wlin+b) + x1
  wmma_mm<128, 4>(sX, 128, Wlin, rowBase, colBase0, acc);
#pragma unroll
  for (int t = 0; t < 4; ++t) {
    int col = colBase0 + t * 16 + m; float bc = b_lin[col];
#pragma unroll
    for (int r = 0; r < 8; ++r) {
      int row = rowBase + r + rh;
      float v = silu_f(acc[t][r] + bc) + x1[(size_t)(eBase + row) * 128 + col];
      sT[row * 128 + col] = f2bf(v);
    }
  }
  __syncthreads();

  // 2x residual blocks (after skip): sX = silu(sT@W1+b1); sT += silu(sX@W2+b2)
#pragma unroll 1
  for (int blk = 0; blk < 2; ++blk) {
    const unsigned short* W1 = Wra1 + blk * 16384;
    const unsigned short* W2 = Wra2 + blk * 16384;
    const float* bb1 = ra_b1 + blk * 128;
    const float* bb2 = ra_b2 + blk * 128;

    wmma_mm<128, 4>(sT, 128, W1, rowBase, colBase0, acc);
#pragma unroll
    for (int t = 0; t < 4; ++t) {
      int col = colBase0 + t * 16 + m; float bc = bb1[col];
#pragma unroll
      for (int r = 0; r < 8; ++r) {
        int row = rowBase + r + rh;
        sX[row * 128 + col] = f2bf(silu_f(acc[t][r] + bc));
      }
    }
    __syncthreads();
    wmma_mm<128, 4>(sX, 128, W2, rowBase, colBase0, acc);
#pragma unroll
    for (int t = 0; t < 4; ++t) {
      int col = colBase0 + t * 16 + m; float bc = bb2[col];
#pragma unroll
      for (int r = 0; r < 8; ++r) {
        int row = rowBase + r + rh;
        int p = row * 128 + col;
        sT[p] = f2bf(bf2f(sT[p]) + silu_f(acc[t][r] + bc));
      }
    }
    __syncthreads();
  }

  // outputs: p1 = sT; p2 = (sbf0 @ w_sbf^T) * p1
  for (int idx = tid; idx < 64 * 128; idx += 256) {
    int e = idx >> 7, c = idx & 127;
    float p1 = bf2f(sT[idx]);
    p1out[(size_t)(eBase + e) * 128 + c] = p1;
    const float* sr = sS0 + e * 42;
    const float* wr = w_sbf + c * 42;
    float s = 0.f;
    for (int k = 0; k < 42; ++k) s += sr[k] * wr[k];
    p2out[(size_t)(eBase + e) * 128 + c] = s * p1;
  }
}

// ---------------------------------------------------------------------------
// Launch
// ---------------------------------------------------------------------------
extern "C" void kernel_launch(void* const* d_in, const int* in_sizes, int n_in,
                              void* d_out, int out_size, void* d_ws, size_t ws_size,
                              hipStream_t stream) {
  const float* x1     = (const float*)d_in[0];
  const float* sbf0   = (const float*)d_in[1];
  const float* ta     = (const float*)d_in[2];
  const int*   pidx   = (const int*)d_in[3];
  const float* w_p2   = (const float*)d_in[4];
  const float* b_p2   = (const float*)d_in[5];
  const float* w_p4   = (const float*)d_in[6];
  const float* b_p4   = (const float*)d_in[7];
  const float* w_sbf1 = (const float*)d_in[8];
  const float* w_sbf2 = (const float*)d_in[9];
  const float* w_t1   = (const float*)d_in[10];
  const float* w_t2   = (const float*)d_in[11];
  const float* w_down = (const float*)d_in[12];
  const float* w_up   = (const float*)d_in[13];
  const float* rb_w1  = (const float*)d_in[14];
  const float* rb_b1  = (const float*)d_in[15];
  const float* rb_w2  = (const float*)d_in[16];
  const float* rb_b2  = (const float*)d_in[17];
  const float* w_lin  = (const float*)d_in[18];
  const float* b_lin  = (const float*)d_in[19];
  const float* ra_w1  = (const float*)d_in[20];
  const float* ra_b1  = (const float*)d_in[21];
  const float* ra_w2  = (const float*)d_in[22];
  const float* ra_b2  = (const float*)d_in[23];
  const float* w_sbf  = (const float*)d_in[24];

  const int E = in_sizes[0] / 128;
  char* ws = (char*)d_ws;
  unsigned short* Wb = (unsigned short*)ws;
  // bf16 weight region (element offsets)
  enum { O_P2 = 0, O_P4 = 16384, O_RB1 = 32768, O_RB2 = 49152, O_LIN = 65536,
         O_RA1 = 81920, O_RA2 = 114688, O_DOWN = 147456, O_UP = 155648, O_END = 163840 };
  unsigned short* xp2 = (unsigned short*)(ws + (size_t)O_END * 2);
  unsigned short* xp4 = xp2 + (size_t)E * 128;

  auto cv = [&](const float* s, int off, int n) {
    k_f2bf<<<(n + 255) / 256, 256, 0, stream>>>(s, Wb + off, n);
  };
  cv(w_p2, O_P2, 16384);
  cv(w_p4, O_P4, 16384);
  cv(rb_w1, O_RB1, 16384);
  cv(rb_w2, O_RB2, 16384);
  cv(w_lin, O_LIN, 16384);
  cv(ra_w1, O_RA1, 32768);
  cv(ra_w2, O_RA2, 32768);
  cv(w_down, O_DOWN, 8192);
  cv(w_up, O_UP, 8192);

  const int nb = E / 64;  // E = 400000 -> 6250 blocks, no tail (EXEC all-ones for WMMA)
  k_stage1<<<nb, 256, 0, stream>>>(x1, sbf0, w_sbf1, w_sbf2, b_p2, b_p4,
                                   Wb + O_P2, Wb + O_P4, Wb + O_DOWN, xp2, xp4);

  float* p1out = (float*)d_out;
  float* p2out = p1out + (size_t)E * 128;
  k_stage2<<<nb, 256, 0, stream>>>(x1, sbf0, ta, pidx, w_t1, w_t2, xp2, xp4,
                                   Wb + O_UP, Wb + O_RB1, Wb + O_RB2, Wb + O_LIN,
                                   Wb + O_RA1, Wb + O_RA2,
                                   rb_b1, rb_b2, b_lin, ra_b1, ra_b2, w_sbf,
                                   p1out, p2out);
}
